// FuncTimeEncoder_6176162972289
// MI455X (gfx1250) — matrix operand order repaired and checked by reading, not compiled, not run on hardware
//
#include <hip/hip_runtime.h>
#include <math.h>

typedef __attribute__((ext_vector_type(16))) _Float16 v16h;
typedef __attribute__((ext_vector_type(8)))  float    v8f;

#define BSZ   65536
#define LL    32
#define NCH   10
#define TT    8
#define EDIM  10
#define KCB   128
#define EMBD  256
#define ZDIM  128
#define KPAD  96      // NC*T = 80 padded to 3x32 for WMMA K-steps
#define ROWP  104     // LDS row pitch in halves (208B = 13*16B, bank-friendly)

// workspace byte offsets
#define WS_W2H   0        // 128*96 f16  (fused weight W2 = mu_w @ fc_w, f16, padded)
#define WS_B2    24576    // 128 f32     (fused bias)
#define WS_CNORM 25088    // 128 f32     (||codebook_k||^2)
#define WS_CNT   25600    // 128 u32     (usage histogram)
#define WS_SUM   26112    // 1 f32       (sum of masked diff^2)
#define WS_VCNT  26116    // 1 u32       (# valid (b,t) positions)

// ---------------- Kernel 1: fold mu∘fc into one GEMM, precompute codebook norms ----
__global__ void prep_kernel(const float* __restrict__ codebook,
                            const float* __restrict__ fc_w,
                            const float* __restrict__ fc_b,
                            const float* __restrict__ mu_w,
                            const float* __restrict__ mu_b,
                            _Float16* __restrict__ W2h,
                            float* __restrict__ b2,
                            float* __restrict__ cnorm) {
  const int n = blockIdx.x;     // 0..127 (ZD rows)
  const int j = threadIdx.x;    // 0..95  (padded K)
  float acc = 0.f;
  if (j < NCH * TT) {
    for (int e = 0; e < EMBD; ++e)
      acc += mu_w[n * EMBD + e] * fc_w[e * (NCH * TT) + j];
  }
  W2h[n * KPAD + j] = (_Float16)acc;   // j in [80,96) -> 0 pad
  if (j == 0) {
    float b = mu_b[n];
    for (int e = 0; e < EMBD; ++e) b += mu_w[n * EMBD + e] * fc_b[e];
    b2[n] = b;
  }
  if (j == 1) {
    float s = 0.f;
    for (int c = 0; c < EDIM; ++c) { float v = codebook[n * EDIM + c]; s += v * v; }
    cnorm[n] = s;
  }
}

// ---------------- Kernel 2: fused conv + VQ + reductions + WMMA GEMM ----------------
__launch_bounds__(128)
__global__ void fused_kernel(const float* __restrict__ pr,
                             const unsigned char* __restrict__ mask,
                             const float* __restrict__ conv_w,
                             const float* __restrict__ conv_b,
                             const float* __restrict__ codebook,
                             const _Float16* __restrict__ W2h,
                             const float* __restrict__ b2,
                             const float* __restrict__ cnorm,
                             float* __restrict__ out,
                             unsigned int* __restrict__ g_cnt,
                             float* __restrict__ g_sum,
                             unsigned int* __restrict__ g_vcnt) {
  __shared__ float        cb_s[KCB * EDIM];     // codebook
  __shared__ float        cn_s[KCB];            // ||c_k||^2
  __shared__ float        cw_s[NCH * 4];
  __shared__ float        cbias_s[NCH];
  __shared__ float        b2_s[ZDIM];
  __shared__ _Float16     W2s[ZDIM * ROWP];     // B matrix (row = output col n, K fast)
  __shared__ _Float16     As[4 * 16 * ROWP];    // per-wave 16-row A tiles (quantized, permuted)
  __shared__ unsigned int hist_s[KCB];
  __shared__ float        red_sum;
  __shared__ unsigned int red_cnt;

  const int tid = threadIdx.x;

  // ---- cooperative staging ----
  for (int i = tid; i < KCB * EDIM; i += 128) cb_s[i] = codebook[i];
  if (tid < KCB) { cn_s[tid] = cnorm[tid]; b2_s[tid] = b2[tid]; hist_s[tid] = 0u; }
  if (tid < NCH * 4) cw_s[tid] = conv_w[tid];
  if (tid < NCH)     cbias_s[tid] = conv_b[tid];
  if (tid == 0) { red_sum = 0.f; red_cnt = 0u; }
  {
    const unsigned int* W2d = (const unsigned int*)W2h;   // 2 halves per dword
    for (int d = tid; d < ZDIM * (KPAD / 2); d += 128) {
      int n = d / (KPAD / 2), j2 = d % (KPAD / 2);
      *(unsigned int*)&W2s[n * ROWP + j2 * 2] = W2d[d];
    }
  }
  __syncthreads();

  const int wave  = tid >> 5;
  const int lane  = tid & 31;
  const int bbase = blockIdx.x * 64 + wave * 16;   // 16 samples per wave

  // ---- conv + VQ: 4 (row,t) pairs per lane (16 rows x 8 t = 128 pairs / wave) ----
  float lsum = 0.f; unsigned int lcnt = 0u;
  for (int p = 0; p < 4; ++p) {
    const int q = lane * 4 + p;        // 0..127
    const int m = q >> 3, t = q & 7;
    const int b = bbase + m;
    const float4 x = *(const float4*)(pr + (size_t)b * LL + t * 4);
    float h[EDIM];
    #pragma unroll
    for (int c = 0; c < EDIM; ++c) {
      float a = cbias_s[c] + x.x * cw_s[c*4+0] + x.y * cw_s[c*4+1]
                           + x.z * cw_s[c*4+2] + x.w * cw_s[c*4+3];
      h[c] = a > 0.f ? a : 0.f;
    }
    // argmin_k ||h - c_k||^2  ==  argmin_k (||c_k||^2 - 2 h.c_k)
    int best = 0; float bestd = 3.4e38f;
    for (int k = 0; k < KCB; ++k) {
      float dot = 0.f;
      #pragma unroll
      for (int c = 0; c < EDIM; ++c) dot += h[c] * cb_s[k * EDIM + c];
      float d2 = cn_s[k] - 2.f * dot;
      if (d2 < bestd) { bestd = d2; best = k; }
    }
    atomicAdd(&hist_s[best], 1u);

    _Float16* arow = &As[(wave * 16 + m) * ROWP];
    float dsum = 0.f;
    #pragma unroll
    for (int c = 0; c < EDIM; ++c) {
      float qv = cb_s[best * EDIM + c];
      arow[c * 8 + t] = (_Float16)qv;     // permuted layout j = c*8 + t
      float df = qv - h[c];
      dsum += df * df;
    }
    if (mask[(size_t)b * TT + t] == 0) { lsum += dsum; lcnt++; }
    if (t == 0) {                         // zero K-pad region [80,96) once per row
      uint4 z = {0u, 0u, 0u, 0u};
      *(uint4*)&arow[80] = z;
      *(uint4*)&arow[88] = z;
    }
  }
  atomicAdd(&red_sum, lsum);
  atomicAdd(&red_cnt, lcnt);
  __syncthreads();

  if (tid < KCB) atomicAdd(&g_cnt[tid], hist_s[tid]);
  if (tid == 0) { atomicAdd(g_sum, red_sum); atomicAdd(g_vcnt, red_cnt); }

  // ---- WMMA GEMM: (16 x 96) A-tile  @  (96 x 128) W2^T, f32 accum, fused bias ----
  union V16 { v16h v; uint4 q[2]; };
  const int half = (lane < 16) ? 0 : 1;
  const int mrow = lane & 15;

  V16 af[3];
  #pragma unroll
  for (int kt = 0; kt < 3; ++kt) {
    // A frag (16x32 f16): lane m=row; halves0-7: K = 32kt + 8*half + i,
    //                     halves8-15: K = 32kt + 16 + 8*half + i
    const _Float16* base = &As[(wave * 16 + mrow) * ROWP + kt * 32];
    af[kt].q[0] = *(const uint4*)(base + half * 8);
    af[kt].q[1] = *(const uint4*)(base + 16 + half * 8);
  }
  #pragma unroll
  for (int nt = 0; nt < 8; ++nt) {
    const int ncol = nt * 16 + mrow;           // C/D column N = lane%16
    const float bias = b2_s[ncol];
    v8f acc = {bias, bias, bias, bias, bias, bias, bias, bias};
    #pragma unroll
    for (int kt = 0; kt < 3; ++kt) {
      // B frag (32x16 f16): column = lane%16; halves0-7: K = 8*half + i, halves8-15: K = 16 + 8*half + i
      V16 bf;
      const _Float16* bp = &W2s[ncol * ROWP + kt * 32 + half * 8];
      bf.q[0] = *(const uint4*)(bp);
      bf.q[1] = *(const uint4*)(bp + 16);
      acc = __builtin_amdgcn_wmma_f32_16x16x32_f16(false, af[kt].v, false, bf.v,
                                                   (short)0, acc, false, false);
    }
    #pragma unroll
    for (int e = 0; e < 8; ++e) {
      const int mr = e + (half ? 8 : 0);       // C layout: M = vgpr + 8*(lane>=16)
      out[(size_t)(bbase + mr) * ZDIM + ncol] = acc[e];
    }
  }
}

// ---------------- Kernel 3: scalars (commit loss, perplexity) -----------------------
__global__ void finalize_kernel(const unsigned int* __restrict__ g_cnt,
                                const float* __restrict__ g_sum,
                                const unsigned int* __restrict__ g_vcnt,
                                float* __restrict__ out_scalars) {
  __shared__ float ssum;
  const int tid = threadIdx.x;
  if (tid == 0) ssum = 0.f;
  __syncthreads();
  if (tid < KCB) {
    float avg = (float)g_cnt[tid] / (float)(BSZ * TT);
    atomicAdd(&ssum, avg * logf(avg + 1e-10f));
  }
  __syncthreads();
  if (tid == 0) {
    out_scalars[0] = 0.25f * (*g_sum) / ((float)(*g_vcnt) * (float)EDIM); // commit loss
    out_scalars[1] = expf(-ssum);                                          // perplexity
  }
}

extern "C" void kernel_launch(void* const* d_in, const int* in_sizes, int n_in,
                              void* d_out, int out_size, void* d_ws, size_t ws_size,
                              hipStream_t stream) {
  const float*         pr       = (const float*)d_in[0];
  const unsigned char* maskp    = (const unsigned char*)d_in[1];  // bool array
  const float*         conv_w   = (const float*)d_in[2];
  const float*         conv_b   = (const float*)d_in[3];
  const float*         codebook = (const float*)d_in[4];
  const float*         fc_w     = (const float*)d_in[5];
  const float*         fc_b     = (const float*)d_in[6];
  const float*         mu_w     = (const float*)d_in[7];
  const float*         mu_b     = (const float*)d_in[8];

  char* ws = (char*)d_ws;
  _Float16*     W2h   = (_Float16*)(ws + WS_W2H);
  float*        b2    = (float*)(ws + WS_B2);
  float*        cnorm = (float*)(ws + WS_CNORM);
  unsigned int* cnt   = (unsigned int*)(ws + WS_CNT);
  float*        gsum  = (float*)(ws + WS_SUM);
  unsigned int* gvcnt = (unsigned int*)(ws + WS_VCNT);

  float* out = (float*)d_out;

  // zero the accumulators (counts + sum + valid count) each call
  hipMemsetAsync(ws + WS_CNT, 0, KCB * 4 + 8, stream);

  prep_kernel<<<KCB, KPAD, 0, stream>>>(codebook, fc_w, fc_b, mu_w, mu_b, W2h, b2, cnorm);

  fused_kernel<<<BSZ / 64, 128, 0, stream>>>(pr, maskp, conv_w, conv_b, codebook,
                                             W2h, b2, cnorm,
                                             out, cnt, gsum, gvcnt);

  finalize_kernel<<<1, 128, 0, stream>>>(cnt, gsum, gvcnt, out + (size_t)BSZ * ZDIM);
}
